// TI_GATNoEdgeAttrNodeProj_39479339384969
// MI455X (gfx1250) — matrix operand order repaired and checked by reading, not compiled, not run on hardware
//
#include <hip/hip_runtime.h>
#include <math.h>

typedef __attribute__((ext_vector_type(16))) _Float16 v16h;
typedef __attribute__((ext_vector_type(8)))  float    v8f;

#define NN 65536
#define EE 1048576
#define HID 64
#define HEADS 4
#define GDIM 16
#define PROJ 8
#define BATCHG 512
#define OUTD 128

// ---------------------------------------------------------------------------
// Templated WMMA GEMM: C[M x NCOLS] = A[M x K] @ W[K x NCOLS] (+bias) (+relu)
// Block = 128 threads (4 waves). Each wave computes one 16x16 tile.
// Block tile: 16 rows x 64 cols. grid = (M/16, NCOLS/64).
// K-chunk of 64 (32 for K=16) staged in LDS, f32 -> f16, f32 accumulate.
// Compile-time K folds all bounds guards but keeps the chunk loop rolled (unroll 1).
// ---------------------------------------------------------------------------
template <int K, int NCOLS, bool HASBIAS, bool RELU>
__global__ __launch_bounds__(128)
void k_gemm(const float* __restrict__ A, const float* __restrict__ W,
            const float* __restrict__ bias, float* __restrict__ C) {
  constexpr int KC = (K >= 64) ? 64 : 32;  // staged K-chunk
  __shared__ float As[16][KC + 1];         // +1 pad: conflict-free column reads
  __shared__ float Ws[KC][64];

  const int lane = threadIdx.x & 31;
  const int wave = threadIdx.x >> 5;
  const int r    = lane & 15;
  const bool hi  = lane >= 16;
  const int rowBase = blockIdx.x * 16;
  const int colBlk  = blockIdx.y * 64;

  v8f acc = {};

#pragma unroll 1
  for (int kc = 0; kc < K; kc += KC) {
    // stage A chunk (16 x KC), coalesced
#pragma unroll
    for (int t = threadIdx.x; t < 16 * KC; t += 128) {
      int rr = t / KC, kk = t % KC, kg = kc + kk;
      As[rr][kk] = (kg < K) ? A[(size_t)(rowBase + rr) * K + kg] : 0.0f;
    }
    // stage W chunk (KC x 64), coalesced
#pragma unroll
    for (int t = threadIdx.x; t < KC * 64; t += 128) {
      int kk = t >> 6, cc = t & 63, kg = kc + kk;
      Ws[kk][cc] = (kg < K) ? W[(size_t)kg * NCOLS + colBlk + cc] : 0.0f;
    }
    __syncthreads();

#pragma unroll
    for (int s = 0; s < KC; s += 32) {
      // A fragment: lane<16 -> row r, K {0..7,16..23}; lane>=16 -> K {8..15,24..31}
      // B fragment: lane<16 -> K {0..7,16..23}; lane>=16 -> K {8..15,24..31}; col = wave*16 + r
      v16h af, bf;
#pragma unroll
      for (int i = 0; i < 16; ++i) {
        int la = s + ((i >> 3) << 4) + (hi ? 8 : 0) + (i & 7);
        af[i] = (_Float16)As[r][la];
        bf[i] = (_Float16)Ws[la][wave * 16 + r];
      }
      acc = __builtin_amdgcn_wmma_f32_16x16x32_f16(
          false, af, false, bf, (short)0, acc, false, false);
    }
    __syncthreads();
  }

  // C/D layout: lane -> col = wave*16 + (lane&15); VGPR j -> row j + 8*(lane>=16)
  const int col = colBlk + wave * 16 + r;
#pragma unroll
  for (int j = 0; j < 8; ++j) {
    int m = rowBase + j + (hi ? 8 : 0);
    float v = acc[j];
    if (HASBIAS) v += bias[col];
    if (RELU) v = fmaxf(v, 0.0f);
    C[(size_t)m * NCOLS + col] = v;
  }
}

// ---------------------------------------------------------------------------
// Per-node attention logits: al_src/al_dst [N,4] = sum_d h[n,h,d]*a[h,d]
// ---------------------------------------------------------------------------
__global__ __launch_bounds__(256)
void k_node_al(const float* __restrict__ h, const float* __restrict__ a_src,
               const float* __restrict__ a_dst, float* __restrict__ als,
               float* __restrict__ ald) {
  int n = blockIdx.x * blockDim.x + threadIdx.x;
  if (n >= NN) return;
  const float* hr = h + (size_t)n * HID;
#pragma unroll
  for (int hh = 0; hh < HEADS; ++hh) {
    float s0 = 0.0f, s1 = 0.0f;
#pragma unroll
    for (int d = 0; d < GDIM; ++d) {
      float hv = hr[hh * GDIM + d];
      s0 += hv * a_src[hh * GDIM + d];
      s1 += hv * a_dst[hh * GDIM + d];
    }
    als[n * HEADS + hh] = s0;
    ald[n * HEADS + hh] = s1;
  }
}

// ---------------------------------------------------------------------------
// Init: zero aggregation buffer, m=-inf, s=0
// ---------------------------------------------------------------------------
__global__ __launch_bounds__(256)
void k_init(float* __restrict__ agg, float* __restrict__ m, float* __restrict__ s) {
  int i = blockIdx.x * blockDim.x + threadIdx.x;
  if (i < NN * HID) agg[i] = 0.0f;
  if (i < NN * HEADS) { m[i] = -__builtin_inff(); s[i] = 0.0f; }
}

__device__ __forceinline__ float leaky(float v) {
  return v > 0.0f ? v : 0.2f * v;
}

// float atomic max via int ordering trick (IEEE-754 monotone bit patterns)
__device__ __forceinline__ void atomicMaxF(float* addr, float val) {
  if (val >= 0.0f)
    atomicMax((int*)addr, __float_as_int(val));
  else
    atomicMin((unsigned int*)addr, (unsigned int)__float_as_int(val));
}

// ---------------------------------------------------------------------------
// Edge pass 1: per (edge,head) leaky logit -> segment max over dst
// ---------------------------------------------------------------------------
__global__ __launch_bounds__(256)
void k_edge_max(const int* __restrict__ src, const int* __restrict__ dst,
                const float* __restrict__ als, const float* __restrict__ ald,
                float* __restrict__ m) {
  int i = blockIdx.x * blockDim.x + threadIdx.x;
  if (i >= EE * HEADS) return;
  int e = i >> 2, hh = i & 3;
  int sN = src[e], dN = dst[e];
  float v = leaky(als[sN * HEADS + hh] + ald[dN * HEADS + hh]);
  atomicMaxF(&m[dN * HEADS + hh], v);
}

// ---------------------------------------------------------------------------
// Edge pass 2: p = exp(e - m[dst]) -> segment sum over dst (e recomputed)
// ---------------------------------------------------------------------------
__global__ __launch_bounds__(256)
void k_edge_sum(const int* __restrict__ src, const int* __restrict__ dst,
                const float* __restrict__ als, const float* __restrict__ ald,
                const float* __restrict__ m, float* __restrict__ s) {
  int i = blockIdx.x * blockDim.x + threadIdx.x;
  if (i >= EE * HEADS) return;
  int e = i >> 2, hh = i & 3;
  int sN = src[e], dN = dst[e];
  float v = leaky(als[sN * HEADS + hh] + ald[dN * HEADS + hh]);
  float p = __expf(v - m[dN * HEADS + hh]);
  atomicAdd(&s[dN * HEADS + hh], p);
}

// ---------------------------------------------------------------------------
// Edge pass 3: alpha = p / (s+eps); agg[dst] += alpha * h[src] (16 dims/head)
// ---------------------------------------------------------------------------
__global__ __launch_bounds__(256)
void k_edge_agg(const int* __restrict__ src, const int* __restrict__ dst,
                const float* __restrict__ als, const float* __restrict__ ald,
                const float* __restrict__ m, const float* __restrict__ s,
                const float* __restrict__ h, float* __restrict__ agg) {
  int i = blockIdx.x * blockDim.x + threadIdx.x;
  if (i >= EE * HEADS) return;
  int e = i >> 2, hh = i & 3;
  int sN = src[e], dN = dst[e];
  float v = leaky(als[sN * HEADS + hh] + ald[dN * HEADS + hh]);
  float p = __expf(v - m[dN * HEADS + hh]);
  float alpha = p / (s[dN * HEADS + hh] + 1e-16f);

  const float4* hs = (const float4*)(h + (size_t)sN * HID + hh * GDIM);
  float* ag = agg + (size_t)dN * HID + hh * GDIM;
  float4 h0 = hs[0], h1 = hs[1], h2 = hs[2], h3 = hs[3];
  atomicAdd(ag + 0,  alpha * h0.x); atomicAdd(ag + 1,  alpha * h0.y);
  atomicAdd(ag + 2,  alpha * h0.z); atomicAdd(ag + 3,  alpha * h0.w);
  atomicAdd(ag + 4,  alpha * h1.x); atomicAdd(ag + 5,  alpha * h1.y);
  atomicAdd(ag + 6,  alpha * h1.z); atomicAdd(ag + 7,  alpha * h1.w);
  atomicAdd(ag + 8,  alpha * h2.x); atomicAdd(ag + 9,  alpha * h2.y);
  atomicAdd(ag + 10, alpha * h2.z); atomicAdd(ag + 11, alpha * h2.w);
  atomicAdd(ag + 12, alpha * h3.x); atomicAdd(ag + 13, alpha * h3.y);
  atomicAdd(ag + 14, alpha * h3.z); atomicAdd(ag + 15, alpha * h3.w);
}

// ---------------------------------------------------------------------------
// x = relu(agg + b)  (in place on agg)
// ---------------------------------------------------------------------------
__global__ __launch_bounds__(256)
void k_bias_relu(float* __restrict__ agg, const float* __restrict__ b) {
  int i = blockIdx.x * blockDim.x + threadIdx.x;
  if (i >= NN * HID) return;
  agg[i] = fmaxf(agg[i] + b[i & (HID - 1)], 0.0f);
}

// ---------------------------------------------------------------------------
// Projection: p[n,8] = relu(h[n,:64] @ Wp[64,8] + bp)
// ---------------------------------------------------------------------------
__global__ __launch_bounds__(256)
void k_proj(const float* __restrict__ h, const float* __restrict__ Wp,
            const float* __restrict__ bp, float* __restrict__ p) {
  int n = blockIdx.x * blockDim.x + threadIdx.x;
  if (n >= NN) return;
  const float* hr = h + (size_t)n * HID;
  float acc[PROJ];
#pragma unroll
  for (int o = 0; o < PROJ; ++o) acc[o] = bp[o];
  for (int k = 0; k < HID; ++k) {
    float hv = hr[k];
#pragma unroll
    for (int o = 0; o < PROJ; ++o) acc[o] += hv * Wp[k * PROJ + o];
  }
#pragma unroll
  for (int o = 0; o < PROJ; ++o) p[(size_t)n * PROJ + o] = fmaxf(acc[o], 0.0f);
}

// ---------------------------------------------------------------------------
// Host-side orchestration
// ---------------------------------------------------------------------------
template <int K>
static void run_gat_layer(const float* xin, const float* W,
                          const float* a_src, const float* a_dst, const float* b,
                          float* hbuf, float* agg, float* als, float* ald,
                          float* mbuf, float* sbuf, const int* src, const int* dst,
                          hipStream_t stream) {
  k_gemm<K, HID, false, false><<<dim3(NN / 16, 1), 128, 0, stream>>>(xin, W, nullptr, hbuf);
  k_node_al<<<NN / 256, 256, 0, stream>>>(hbuf, a_src, a_dst, als, ald);
  k_init<<<(NN * HID + 255) / 256, 256, 0, stream>>>(agg, mbuf, sbuf);
  const int eh = EE * HEADS;
  k_edge_max<<<(eh + 255) / 256, 256, 0, stream>>>(src, dst, als, ald, mbuf);
  k_edge_sum<<<(eh + 255) / 256, 256, 0, stream>>>(src, dst, als, ald, mbuf, sbuf);
  k_edge_agg<<<(eh + 255) / 256, 256, 0, stream>>>(src, dst, als, ald, mbuf, sbuf, hbuf, agg);
  k_bias_relu<<<(NN * HID + 255) / 256, 256, 0, stream>>>(agg, b);
}

extern "C" void kernel_launch(void* const* d_in, const int* in_sizes, int n_in,
                              void* d_out, int out_size, void* d_ws, size_t ws_size,
                              hipStream_t stream) {
  const float* x   = (const float*)d_in[0];
  const int*   ei  = (const int*)  d_in[1];
  // d_in[2] = batch (implicit via reshape; unused)
  const float* W0  = (const float*)d_in[3];
  const float* as0 = (const float*)d_in[4];
  const float* ad0 = (const float*)d_in[5];
  const float* b0  = (const float*)d_in[6];
  const float* W1  = (const float*)d_in[7];
  const float* as1 = (const float*)d_in[8];
  const float* ad1 = (const float*)d_in[9];
  const float* b1  = (const float*)d_in[10];
  const float* W2  = (const float*)d_in[11];
  const float* as2 = (const float*)d_in[12];
  const float* ad2 = (const float*)d_in[13];
  const float* b2  = (const float*)d_in[14];
  const float* Wp  = (const float*)d_in[15];
  const float* bp  = (const float*)d_in[16];
  const float* Wf  = (const float*)d_in[17];
  const float* bfv = (const float*)d_in[18];

  const int* src = ei;
  const int* dst = ei + EE;

  float* wsf  = (float*)d_ws;
  float* xbuf = wsf;                      // N*64 : layer input / aggregation (ping)
  float* hbuf = xbuf + (size_t)NN * HID;  // N*64 : h (also holds proj output)
  float* als  = hbuf + (size_t)NN * HID;  // N*4
  float* ald  = als + NN * HEADS;         // N*4
  float* mbuf = ald + NN * HEADS;         // N*4
  float* sbuf = mbuf + NN * HEADS;        // N*4

  // Layer 0 (K=16 from raw x), aggregated output lands in xbuf
  run_gat_layer<16>(x,    W0, as0, ad0, b0, hbuf, xbuf, als, ald, mbuf, sbuf, src, dst, stream);
  // Layers 1-2 (K=64): xbuf -> hbuf -> agg back into xbuf
  run_gat_layer<64>(xbuf, W1, as1, ad1, b1, hbuf, xbuf, als, ald, mbuf, sbuf, src, dst, stream);
  run_gat_layer<64>(xbuf, W2, as2, ad2, b2, hbuf, xbuf, als, ald, mbuf, sbuf, src, dst, stream);

  // Projection: xbuf [N,64] -> hbuf [N,8]
  k_proj<<<NN / 256, 256, 0, stream>>>(xbuf, Wp, bp, hbuf);

  // Final classifier: view hbuf as [512, 1024]; out[512,128] = P @ Wf + bf (WMMA)
  k_gemm<NN * PROJ / BATCHG /*1024*/, OUTD, true, false>
      <<<dim3(BATCHG / 16, OUTD / 64), 128, 0, stream>>>(hbuf, Wf, bfv, (float*)d_out);
}
